// FeatureMap_3195455668524
// MI455X (gfx1250) — compile-verified, matrix operand order checked by reading, not run on hardware
//
#include <hip/hip_runtime.h>
#include <hip/hip_bf16.h>

typedef _Float16 h4  __attribute__((ext_vector_type(4)));
typedef _Float16 h8  __attribute__((ext_vector_type(8)));
typedef _Float16 h16 __attribute__((ext_vector_type(16)));
typedef float    v8f __attribute__((ext_vector_type(8)));

#define NNODES 100000
#define NEDGES 3200000
#define INDIM  32
#define HID    128
#define EPS    1e-5f

// ---------------------------------------------------------------- zero fill
__global__ __launch_bounds__(256) void zero_kernel(float* p, long long count) {
    long long i = (long long)blockIdx.x * blockDim.x + threadIdx.x;
    long long stride = (long long)gridDim.x * blockDim.x;
    for (; i < count; i += stride) p[i] = 0.0f;
}

// ------------------------------------------------- per-channel sum / sumsq
__global__ __launch_bounds__(256) void colstats_kernel(const float* __restrict__ x,
                                                       float* __restrict__ stats, int n) {
    __shared__ float ss[64];
    int t = threadIdx.x;
    if (t < 64) ss[t] = 0.0f;
    __syncthreads();
    int c = t & 31;                       // channel
    int rpb = blockDim.x >> 5;            // rows handled per block per step (8)
    float s = 0.0f, q = 0.0f;
    for (int r = blockIdx.x * rpb + (t >> 5); r < n; r += gridDim.x * rpb) {
        float v = x[(long long)r * INDIM + c];
        s += v; q += v * v;
    }
    atomicAdd(&ss[c], s);
    atomicAdd(&ss[32 + c], q);
    __syncthreads();
    if (t < 64) atomicAdd(&stats[t], ss[t]);
}

// ------------------------- normalize x and write into Hcat[:, 0:32] (hop 0)
__global__ __launch_bounds__(256) void normalize_kernel(const float* __restrict__ x,
                                                        const float* __restrict__ stats,
                                                        float* __restrict__ hcat, int n) {
    long long idx = (long long)blockIdx.x * blockDim.x + threadIdx.x;
    if (idx >= (long long)n * INDIM) return;
    int c = (int)(idx & 31);
    long long r = idx >> 5;
    float invN = 1.0f / (float)n;
    float mean = stats[c] * invN;
    float var  = stats[32 + c] * invN - mean * mean;
    float v = (x[idx] - mean) * rsqrtf(var + EPS);
    hcat[r * HID + c] = v;
}

// ---------------------------------------------------------------- degrees
__global__ __launch_bounds__(256) void deg_kernel(const long long* __restrict__ ei,
                                                  const float* __restrict__ attr,
                                                  float* __restrict__ deg, int e) {
    int i = blockIdx.x * blockDim.x + threadIdx.x;
    if (i >= e) return;
    long long dst = ei[(long long)NEDGES + i];
    atomicAdd(&deg[dst], attr[i]);
}

__global__ __launch_bounds__(256) void dinv_kernel(float* __restrict__ deg, int n) {
    int i = blockIdx.x * blockDim.x + threadIdx.x;
    if (i >= n) return;
    float d = deg[i];
    deg[i] = (d > 0.0f) ? rsqrtf(d) : 0.0f;
}

__global__ __launch_bounds__(256) void edgenorm_kernel(const long long* __restrict__ ei,
                                                       const float* __restrict__ attr,
                                                       const float* __restrict__ dinv,
                                                       float* __restrict__ enorm, int e) {
    int i = blockIdx.x * blockDim.x + threadIdx.x;
    if (i >= e) return;
    long long src = ei[i];
    long long dst = ei[(long long)NEDGES + i];
    enorm[i] = dinv[src] * attr[i] * dinv[dst];
}

// ------------- one propagation hop: Hcat[:,out] += norm[e] * Hcat[src, in]
// one wave per edge, lane == channel; feature rows stay L2-resident
__global__ __launch_bounds__(256) void spmm_hop_kernel(const long long* __restrict__ ei,
                                                       const float* __restrict__ enorm,
                                                       float* __restrict__ hcat,
                                                       int in_off, int out_off, int e) {
    long long tid = (long long)blockIdx.x * blockDim.x + threadIdx.x;
    long long edge = tid >> 5;
    if (edge >= e) return;
    int lane = threadIdx.x & 31;
    long long src = ei[edge];
    long long dst = ei[(long long)NEDGES + edge];
    float w = enorm[edge];
    float v = w * hcat[src * HID + in_off + lane];
    atomicAdd(&hcat[dst * HID + out_off + lane], v);
}

// --------------------------- C[nrows,128] = act(A[nrows,128] @ W[128,128] + b)
// 256 threads = 8 waves; block tile = 64 rows x 128 cols (4 M-subtiles of 16).
// Wave w owns output cols [16w, 16w+16). B fragments preloaded once per block
// and reused over all 4 M-subtiles. nrows must be a multiple of 16, so each
// M-subtile is either fully valid or fully absent (no per-row guards).
template <bool RELU>
__global__ __launch_bounds__(256) void gemm_bias_act_kernel(const float* __restrict__ A,
                                                            const float* __restrict__ W,
                                                            const float* __restrict__ bias,
                                                            float* __restrict__ C,
                                                            int nrows) {
    __shared__ __align__(16) _Float16 Wt[128 * 136];   // Wt[n*136 + k] = W[k*128 + n]
    __shared__ __align__(16) _Float16 As[64 * 136];    // As[m*136 + k]
    const int t = threadIdx.x;
    const int row0 = blockIdx.x * 64;

    // stage transposed weights: float4 global reads, scalar LDS writes (transpose)
    const float4* W4 = (const float4*)W;
    #pragma unroll
    for (int i = 0; i < 16; ++i) {
        int idx4 = i * 256 + t;                // 4096 float4s
        int base = idx4 << 2;
        int k = base >> 7, n = base & 127;
        float4 wv = W4[idx4];
        Wt[(n + 0) * 136 + k] = (_Float16)wv.x;
        Wt[(n + 1) * 136 + k] = (_Float16)wv.y;
        Wt[(n + 2) * 136 + k] = (_Float16)wv.z;
        Wt[(n + 3) * 136 + k] = (_Float16)wv.w;
    }
    // stage A tile (64 rows x 128): float4 reads -> packed ds_store_b64
    const float4* A4 = (const float4*)A;
    #pragma unroll
    for (int i = 0; i < 8; ++i) {
        int idx4 = i * 256 + t;                // 2048 float4s
        int base = idx4 << 2;
        int m = base >> 7, k = base & 127;
        int row = row0 + m;
        float4 av = (row < nrows) ? A4[(long long)row * 32 + (k >> 2)]
                                  : make_float4(0.f, 0.f, 0.f, 0.f);
        h4 p;
        p[0] = (_Float16)av.x; p[1] = (_Float16)av.y;
        p[2] = (_Float16)av.z; p[3] = (_Float16)av.w;
        *(h4*)(&As[m * 136 + k]) = p;
    }
    __syncthreads();

    const int wave  = t >> 5;                  // output col tile
    const int lane  = t & 31;
    const int mr    = lane & 15;               // A row / B col within tile
    const int half  = lane >> 4;
    const int koff0 = half * 8;                // f16 frag layout: two contiguous runs of 8
    const int koff1 = 16 + half * 8;
    const int ncol  = wave * 16 + mr;

    // preload all B fragments for this wave's column tile (reused 4x)
    h16 bfrag[4];
    const _Float16* brow = &Wt[ncol * 136];
    #pragma unroll
    for (int kt = 0; kt < 4; ++kt) {
        h8 b0 = *(const h8*)(brow + kt * 32 + koff0);
        h8 b1 = *(const h8*)(brow + kt * 32 + koff1);
        bfrag[kt] = __builtin_shufflevector(b0, b1, 0,1,2,3,4,5,6,7,8,9,10,11,12,13,14,15);
    }
    const float bv = bias[ncol];

    #pragma unroll
    for (int sub = 0; sub < 4; ++sub) {
        const int mrow0 = row0 + sub * 16;
        if (mrow0 >= nrows) break;             // whole subtile valid or absent
        const _Float16* arow = &As[(sub * 16 + mr) * 136];
        v8f acc = {};
        #pragma unroll
        for (int kt = 0; kt < 4; ++kt) {
            h8 a0 = *(const h8*)(arow + kt * 32 + koff0);
            h8 a1 = *(const h8*)(arow + kt * 32 + koff1);
            h16 af = __builtin_shufflevector(a0, a1, 0,1,2,3,4,5,6,7,8,9,10,11,12,13,14,15);
            acc = __builtin_amdgcn_wmma_f32_16x16x32_f16(false, af, false, bfrag[kt],
                                                         (short)0, acc, false, false);
        }
        #pragma unroll
        for (int r = 0; r < 8; ++r) {
            int row = mrow0 + r + half * 8;
            float v = acc[r] + bv;
            if (RELU) v = fmaxf(v, 0.0f);
            C[(long long)row * HID + ncol] = v;
        }
    }
}

extern "C" void kernel_launch(void* const* d_in, const int* in_sizes, int n_in,
                              void* d_out, int out_size, void* d_ws, size_t ws_size,
                              hipStream_t stream) {
    const float*     x     = (const float*)d_in[0];
    const long long* ei    = (const long long*)d_in[1];
    const float*     attr  = (const float*)d_in[2];
    const float*     w_tag = (const float*)d_in[3];   // (4,32,128) == row-major [128,128]
    const float*     b_tag = (const float*)d_in[4];
    const float*     w1    = (const float*)d_in[5];
    const float*     b1    = (const float*)d_in[6];
    const float*     w2    = (const float*)d_in[7];
    const float*     b2    = (const float*)d_in[8];
    const float*     w3    = (const float*)d_in[9];
    const float*     b3    = (const float*)d_in[10];
    float*           out   = (float*)d_out;

    const int N = NNODES, E = NEDGES;

    float* ws    = (float*)d_ws;
    float* stats = ws;                                  // 64 floats
    float* deg   = stats + 64;                          // N
    float* hcat  = deg + N;                             // N*128 (contiguous with zero prefix)
    float* enorm = hcat + (long long)N * HID;           // E
    float* buf   = enorm + E;                           // N*128

    // zero stats + deg + Hcat (contiguous prefix)
    long long zcount = 64 + (long long)N + (long long)N * HID;
    zero_kernel<<<4096, 256, 0, stream>>>(ws, zcount);

    // InstanceNorm stats + normalize into hop-0 slice of Hcat
    colstats_kernel<<<1024, 256, 0, stream>>>(x, stats, N);
    normalize_kernel<<<(N * INDIM + 255) / 256, 256, 0, stream>>>(x, stats, hcat, N);

    // gcn_norm
    deg_kernel<<<(E + 255) / 256, 256, 0, stream>>>(ei, attr, deg, E);
    dinv_kernel<<<(N + 255) / 256, 256, 0, stream>>>(deg, N);
    edgenorm_kernel<<<(E + 255) / 256, 256, 0, stream>>>(ei, attr, deg, enorm, E);

    // K=3 propagation hops: slice k-1 -> slice k of Hcat
    long long spmm_threads = (long long)E * 32;
    int spmm_blocks = (int)((spmm_threads + 255) / 256);
    spmm_hop_kernel<<<spmm_blocks, 256, 0, stream>>>(ei, enorm, hcat, 0,  32, E);
    spmm_hop_kernel<<<spmm_blocks, 256, 0, stream>>>(ei, enorm, hcat, 32, 64, E);
    spmm_hop_kernel<<<spmm_blocks, 256, 0, stream>>>(ei, enorm, hcat, 64, 96, E);

    // TAGConv linear (fused over hops) + MLP, all as WMMA GEMMs
    int gblocks = (N + 63) / 64;
    gemm_bias_act_kernel<true ><<<gblocks, 256, 0, stream>>>(hcat, w_tag, b_tag, buf,  N);
    gemm_bias_act_kernel<true ><<<gblocks, 256, 0, stream>>>(buf,  w1,    b1,    hcat, N);
    gemm_bias_act_kernel<true ><<<gblocks, 256, 0, stream>>>(hcat, w2,    b2,    buf,  N);
    gemm_bias_act_kernel<false><<<gblocks, 256, 0, stream>>>(buf,  w3,    b3,    out,  N);
}
